// LHC_57320633532463
// MI455X (gfx1250) — matrix operand order repaired and checked by reading, not compiled, not run on hardware
//
#include <hip/hip_runtime.h>

// ---------------------------------------------------------------------------
// Fused 47-frame particle simulation for MI455X (gfx1250, wave32, WMMA).
// Round 3: nine of ten 1x1-conv matmuls (GRU z/r/q, rule MLP, Wv1; Wv2 is a VALU dot) now run
// on v_wmma_f32_16x16x32_f16, not just the splat. The channel contraction
// (K=32) makes the GRU/MLP chain wave-local (each wave owns 16-particle column
// slices), so gates are applied elementwise on D-fragments and D->B layout
// changes round-trip through the wave's own LDS slice (same-wave DS ordering).
// This removes the 128+ float per-thread arrays that caused the scratch spills
// seen in rounds 1-2, and cuts static VALU massively.
//   LDS: XP 128K + XN 128K + pos 8K + weights 32K ~= 295KB < 320KB/WGP.
//   Barriers: 2 per frame (after KNN-aggregate, after state update).
// ---------------------------------------------------------------------------

#define NB       16
#define NC       32
#define NPART    1024     // W*H = 32*32 particles / grid cells
#define NFR      47
#define KNN      8
#define NTHR     512
#define GSTEP    (2.0f / 31.0f)   // linspace(-1,1,32) step

typedef _Float16 v16h __attribute__((ext_vector_type(16)));
typedef float    v8f  __attribute__((ext_vector_type(8)));

union AFrag { v16h v; _Float16 h[16]; };
union CFrag { v8f  v; float     f[8]; };

#define WMMA(A, B, C) __builtin_amdgcn_wmma_f32_16x16x32_f16(false, (A), false, (B), (short)0, (C), false, false)

// ---- dynamic-LDS layout (floats) ----
#define XP_OFF   0                       // x_part [32][1024]
#define XN_OFF   (XP_OFF + NC * NPART)   // scratch matrix [32][1024] (xn / h)
#define PX_OFF   (XN_OFF + NC * NPART)   // posx [1024]
#define PY_OFF   (PX_OFF + NPART)        // posy [1024]
#define WB_OFF   (PY_OFF + NPART)        // packed weights (7874 floats)
#define SMEM_FLOATS (WB_OFF + 7874)      // 75458 floats = 301,832 B

// weight offsets inside wb
#define O_WR1 0
#define O_BR1 512
#define O_WR2 528
#define O_BR2 1040
#define O_WV1 1072
#define O_BV1 1584
#define O_WV2 1600
#define O_BV2 1632
#define O_WXZ 1634
#define O_WHZ 2658
#define O_BZ  3682
#define O_WXR 3714
#define O_WHR 4738
#define O_BRG 5762
#define O_WXQ 5794
#define O_WHQ 6818
#define O_BQ  7842

__device__ __forceinline__ float frcp(float x)     { return __builtin_amdgcn_rcpf(x); }
__device__ __forceinline__ float fsigmoid(float x) { return frcp(1.0f + __expf(-x)); }
__device__ __forceinline__ float ftanh(float x)    { return 1.0f - 2.0f * frcp(1.0f + __expf(2.0f * x)); }

// B-fragment (32x16): K = channel, N = particle column.  K(h,lane)=h+16*(lane>=16)
__device__ __forceinline__ AFrag loadBfrag(const float* m, int col, int kadd_b) {
    AFrag b;
#pragma unroll
    for (int h = 0; h < 16; ++h) b.h[h] = (_Float16)m[(h + kadd_b) * NPART + col];
    return b;
}
// B-fragment with only K<16 valid (rule-MLP hidden), zero-padded to K=32
__device__ __forceinline__ AFrag loadBfrag16(const float* m, int col, int kadd_b) {
    AFrag b;
#pragma unroll
    for (int h = 0; h < 16; ++h) {
        int k = h + kadd_b;
        b.h[h] = (k < 16) ? (_Float16)m[k * NPART + col] : (_Float16)0.0f;
    }
    return b;
}
// A-fragment (16x32) of a row-major weight W[row][k], rows mrow..mrow+15,
// zero-padded for k >= ldk.  K(h,lane) = h + 16*(lane>=16)
__device__ __forceinline__ AFrag loadAfrag(const float* W, int ldk, int mrow, int l15, int kadd_a) {
    AFrag a;
#pragma unroll
    for (int h = 0; h < 16; ++h) {
        int k = h + kadd_a;
        a.h[h] = (k < ldk) ? (_Float16)W[(mrow + l15) * ldk + k] : (_Float16)0.0f;
    }
    return a;
}
// D/C fragment helpers: row M = mrow + r + rbase, col fixed per lane
__device__ __forceinline__ CFrag biasC(const float* bias, int mrow, int rbase) {
    CFrag c;
#pragma unroll
    for (int r = 0; r < 8; ++r) c.f[r] = bias[mrow + rbase + r];
    return c;
}
__device__ __forceinline__ CFrag loadDfrag(const float* m, int mrow, int col, int rbase) {
    CFrag c;
#pragma unroll
    for (int r = 0; r < 8; ++r) c.f[r] = m[(mrow + rbase + r) * NPART + col];
    return c;
}
__device__ __forceinline__ void storeDfrag(float* m, const CFrag& c, int mrow, int col, int rbase) {
#pragma unroll
    for (int r = 0; r < 8; ++r) m[(mrow + rbase + r) * NPART + col] = c.f[r];
}

__global__ __launch_bounds__(NTHR)
void LHC_57320633532463_kernel(
    const float* __restrict__ x,
    const float* __restrict__ gWr1, const float* __restrict__ gbr1,
    const float* __restrict__ gWr2, const float* __restrict__ gbr2,
    const float* __restrict__ gWv1, const float* __restrict__ gbv1,
    const float* __restrict__ gWv2, const float* __restrict__ gbv2,
    const float* __restrict__ gWxz, const float* __restrict__ gWhz, const float* __restrict__ gbz,
    const float* __restrict__ gWxr, const float* __restrict__ gWhr, const float* __restrict__ gbrg,
    const float* __restrict__ gWxq, const float* __restrict__ gWhq, const float* __restrict__ gbq,
    float* __restrict__ out)
{
    extern __shared__ float smem[];
    float* xpm = smem + XP_OFF;
    float* xnm = smem + XN_OFF;
    float* px_ = smem + PX_OFF;
    float* py_ = smem + PY_OFF;
    float* wb  = smem + WB_OFF;

    const int tid = threadIdx.x;
    const int b   = blockIdx.x;

    // ---- stage weights + initial state into LDS ----
    auto cp = [&](int off, const float* src, int n) {
        for (int t = tid; t < n; t += NTHR) wb[off + t] = src[t];
    };
    cp(O_WR1, gWr1, 512);  cp(O_BR1, gbr1, 16);
    cp(O_WR2, gWr2, 512);  cp(O_BR2, gbr2, 32);
    cp(O_WV1, gWv1, 512);  cp(O_BV1, gbv1, 16);
    cp(O_WV2, gWv2, 32);   cp(O_BV2, gbv2, 2);
    cp(O_WXZ, gWxz, 1024); cp(O_WHZ, gWhz, 1024); cp(O_BZ,  gbz,  32);
    cp(O_WXR, gWxr, 1024); cp(O_WHR, gWhr, 1024); cp(O_BRG, gbrg, 32);
    cp(O_WXQ, gWxq, 1024); cp(O_WHQ, gWhq, 1024); cp(O_BQ,  gbq,  32);

    for (int t = tid; t < NC * NPART; t += NTHR)
        xpm[t] = x[(size_t)b * NC * NPART + t];
    for (int n = tid; n < NPART; n += NTHR) {
        // uniform grid: n = w*32 + h, channel0 = px(w), channel1 = py(h)
        px_[n] = -1.0f + (float)(n >> 5) * GSTEP;
        py_[n] = -1.0f + (float)(n & 31) * GSTEP;
    }
    __syncthreads();

    const int lane   = tid & 31;
    const int wave   = tid >> 5;
    const int l15    = lane & 15;
    const int kadd_a = (lane & 16) ? 16 : 0;
    const int kadd_b = (lane & 16) ? 16 : 0;
    const int rbase  = (lane & 16) ? 8  : 0;

#pragma unroll 1
    for (int f = 0; f < NFR; ++f) {
        // ========== phase A: KNN + neighbor aggregation (thread=particle) ===
#pragma unroll 1
        for (int p = 0; p < 2; ++p) {
            const int i = tid + p * NTHR;
            const float pix = px_[i], piy = py_[i];

            float nd[KNN]; int ni[KNN];
#pragma unroll
            for (int k = 0; k < KNN; ++k) { nd[k] = 1e30f; ni[k] = 0; }
#pragma unroll 1
            for (int j = 0; j < NPART; ++j) {      // includes self (d=0), as ref
                float dx = pix - px_[j], dy = piy - py_[j];
                float d  = dx * dx + dy * dy;
                if (d < nd[KNN - 1]) {
                    nd[KNN - 1] = d; ni[KNN - 1] = j;
#pragma unroll
                    for (int k = KNN - 1; k > 0; --k) {
                        if (nd[k] < nd[k - 1]) {
                            float td = nd[k]; nd[k] = nd[k - 1]; nd[k - 1] = td;
                            int   ti = ni[k]; ni[k] = ni[k - 1]; ni[k - 1] = ti;
                        }
                    }
                }
            }
            float xn[NC];
#pragma unroll
            for (int c = 0; c < NC; ++c) xn[c] = 0.0f;
#pragma unroll
            for (int k = 0; k < KNN; ++k) {
                float w = frcp(nd[k] + 1e-8f);
                int   j = ni[k];
#pragma unroll
                for (int c = 0; c < NC; ++c) xn[c] += xpm[c * NPART + j] * w;
            }
#pragma unroll
            for (int c = 0; c < NC; ++c) xnm[c * NPART + i] = xn[c] * (1.0f / (float)KNN);
        }
        __syncthreads();   // XN complete

        // ========== phase B: GRU + MLPs on WMMA (wave-local column slices) ==
#pragma unroll 1
        for (int it = 0; it < 4; ++it) {
            const int col = (wave * 4 + it) * 16 + l15;

            AFrag XNb = loadBfrag(xnm, col, kadd_b);
            AFrag XPb = loadBfrag(xpm, col, kadd_b);

            // z = sigmoid(Wxz xn + Whz xp + bz)
            CFrag z0 = biasC(wb + O_BZ, 0, rbase), z1 = biasC(wb + O_BZ, 16, rbase);
            {
                AFrag w = loadAfrag(wb + O_WHZ, 32, 0,  l15, kadd_a); z0.v = WMMA(w.v, XPb.v, z0.v);
                w       = loadAfrag(wb + O_WHZ, 32, 16, l15, kadd_a); z1.v = WMMA(w.v, XPb.v, z1.v);
                w       = loadAfrag(wb + O_WXZ, 32, 0,  l15, kadd_a); z0.v = WMMA(w.v, XNb.v, z0.v);
                w       = loadAfrag(wb + O_WXZ, 32, 16, l15, kadd_a); z1.v = WMMA(w.v, XNb.v, z1.v);
            }
#pragma unroll
            for (int r = 0; r < 8; ++r) { z0.f[r] = fsigmoid(z0.f[r]); z1.f[r] = fsigmoid(z1.f[r]); }

            // r = sigmoid(Wxr xn + Whr xp + brg)
            CFrag r0 = biasC(wb + O_BRG, 0, rbase), r1 = biasC(wb + O_BRG, 16, rbase);
            {
                AFrag w = loadAfrag(wb + O_WHR, 32, 0,  l15, kadd_a); r0.v = WMMA(w.v, XPb.v, r0.v);
                w       = loadAfrag(wb + O_WHR, 32, 16, l15, kadd_a); r1.v = WMMA(w.v, XPb.v, r1.v);
                w       = loadAfrag(wb + O_WXR, 32, 0,  l15, kadd_a); r0.v = WMMA(w.v, XNb.v, r0.v);
                w       = loadAfrag(wb + O_WXR, 32, 16, l15, kadd_a); r1.v = WMMA(w.v, XNb.v, r1.v);
            }
            CFrag xo0 = loadDfrag(xpm, 0, col, rbase), xo1 = loadDfrag(xpm, 16, col, rbase);
            CFrag rx0, rx1;
#pragma unroll
            for (int r = 0; r < 8; ++r) {
                rx0.f[r] = fsigmoid(r0.f[r]) * xo0.f[r];
                rx1.f[r] = fsigmoid(r1.f[r]) * xo1.f[r];
            }
            // rx overwrites this wave's XP slice (only this wave reads it)
            storeDfrag(xpm, rx0, 0, col, rbase); storeDfrag(xpm, rx1, 16, col, rbase);

            // q = tanh(Wxq xn + Whq (r*xp) + bq)
            CFrag q0 = biasC(wb + O_BQ, 0, rbase), q1 = biasC(wb + O_BQ, 16, rbase);
            {
                AFrag w = loadAfrag(wb + O_WXQ, 32, 0,  l15, kadd_a); q0.v = WMMA(w.v, XNb.v, q0.v);
                w       = loadAfrag(wb + O_WXQ, 32, 16, l15, kadd_a); q1.v = WMMA(w.v, XNb.v, q1.v);
            }
            AFrag RXb = loadBfrag(xpm, col, kadd_b);   // same-wave DS ops are in-order
            {
                AFrag w = loadAfrag(wb + O_WHQ, 32, 0,  l15, kadd_a); q0.v = WMMA(w.v, RXb.v, q0.v);
                w       = loadAfrag(wb + O_WHQ, 32, 16, l15, kadd_a); q1.v = WMMA(w.v, RXb.v, q1.v);
            }
            // nx = (1-z)*xp + z*q
            CFrag nx0, nx1;
#pragma unroll
            for (int r = 0; r < 8; ++r) {
                nx0.f[r] = (1.0f - z0.f[r]) * xo0.f[r] + z0.f[r] * ftanh(q0.f[r]);
                nx1.f[r] = (1.0f - z1.f[r]) * xo1.f[r] + z1.f[r] * ftanh(q1.f[r]);
            }
            storeDfrag(xpm, nx0, 0, col, rbase); storeDfrag(xpm, nx1, 16, col, rbase);

            // rule MLP: nx += relu(Wr2 relu(Wr1 nx + br1) + br2)
            AFrag NXb = loadBfrag(xpm, col, kadd_b);
            CFrag hf = biasC(wb + O_BR1, 0, rbase);
            {
                AFrag w = loadAfrag(wb + O_WR1, 32, 0, l15, kadd_a); hf.v = WMMA(w.v, NXb.v, hf.v);
            }
#pragma unroll
            for (int r = 0; r < 8; ++r) hf.f[r] = fmaxf(hf.f[r], 0.0f);
            storeDfrag(xnm, hf, 0, col, rbase);        // h (16 rows) into XN scratch slice
            AFrag Hb = loadBfrag16(xnm, col, kadd_b);  // K padded 16 -> 32
            CFrag o20 = biasC(wb + O_BR2, 0, rbase), o21 = biasC(wb + O_BR2, 16, rbase);
            {
                AFrag w = loadAfrag(wb + O_WR2, 16, 0,  l15, kadd_a); o20.v = WMMA(w.v, Hb.v, o20.v);
                w       = loadAfrag(wb + O_WR2, 16, 16, l15, kadd_a); o21.v = WMMA(w.v, Hb.v, o21.v);
            }
#pragma unroll
            for (int r = 0; r < 8; ++r) {
                nx0.f[r] += fmaxf(o20.f[r], 0.0f);
                nx1.f[r] += fmaxf(o21.f[r], 0.0f);
            }
            storeDfrag(xpm, nx0, 0, col, rbase); storeDfrag(xpm, nx1, 16, col, rbase);

            // velocity MLP: v = tanh(Wv2 relu(Wv1 nx + bv1) + bv2)/32
            AFrag NX2b = loadBfrag(xpm, col, kadd_b);
            CFrag hv = biasC(wb + O_BV1, 0, rbase);
            {
                AFrag w = loadAfrag(wb + O_WV1, 32, 0, l15, kadd_a); hv.v = WMMA(w.v, NX2b.v, hv.v);
            }
#pragma unroll
            for (int r = 0; r < 8; ++r) hv.f[r] = fmaxf(hv.f[r], 0.0f);
            float s0 = 0.0f, s1 = 0.0f;
#pragma unroll
            for (int r = 0; r < 8; ++r) {
                int row = r + rbase;
                s0 += wb[O_WV2 + row]      * hv.f[r];
                s1 += wb[O_WV2 + 16 + row] * hv.f[r];
            }
            s0 += __shfl_xor(s0, 16);
            s1 += __shfl_xor(s1, 16);
            float vx = ftanh(s0 + wb[O_BV2 + 0]) * (1.0f / 32.0f);
            float vy = ftanh(s1 + wb[O_BV2 + 1]) * (1.0f / 32.0f);
            if (lane < 16) {
                px_[col] = fminf(fmaxf(px_[col] + vx, -1.5f), 1.5f);
                py_[col] = fminf(fmaxf(py_[col] + vy, -1.5f), 1.5f);
            }
        }
        __syncthreads();   // XP = new x_part, pos updated, visible to all

        // ========== phase C: truncated-exp splat via WMMA ===================
        float* op = out + (size_t)(b * NFR + f) * (NC * NPART);
#pragma unroll 1
        for (int pr = 0; pr < 2; ++pr) {
            const int t0   = pr * 32 + wave * 2;
            const int col0 = t0 * 16 + l15;
            const int col1 = col0 + 16;
            const float gx0 = -1.0f + (float)(col0 >> 5) * GSTEP;
            const float gy0 = -1.0f + (float)(col0 & 31) * GSTEP;
            const float gx1 = -1.0f + (float)(col1 >> 5) * GSTEP;
            const float gy1 = -1.0f + (float)(col1 & 31) * GSTEP;

            CFrag a00, a10, a01, a11;
#pragma unroll
            for (int r = 0; r < 8; ++r) { a00.f[r] = 0.f; a10.f[r] = 0.f; a01.f[r] = 0.f; a11.f[r] = 0.f; }
            float hit0 = 0.0f, hit1 = 0.0f;

#pragma unroll 1
            for (int kb = 0; kb < NPART; kb += 32) {
                AFrag fa0, fa1, fb0, fb1;
#pragma unroll
                for (int h = 0; h < 16; ++h) {
                    int ka = kb + h + kadd_a;
                    fa0.h[h] = (_Float16)xpm[l15 * NPART + ka];
                    fa1.h[h] = (_Float16)xpm[(16 + l15) * NPART + ka];
                    int kk = kb + h + kadd_b;
                    float px = px_[kk], py = py_[kk];
                    float dx0 = px - gx0, dy0 = py - gy0;
                    float d0  = dx0 * dx0 + dy0 * dy0;
                    float v0  = (d0 <= 0.8f) ? __expf(-5.0f * d0) : 0.0f;
                    hit0 += (v0 > 0.8f) ? 1.0f : 0.0f;
                    fb0.h[h] = (_Float16)v0;
                    float dx1 = px - gx1, dy1 = py - gy1;
                    float d1  = dx1 * dx1 + dy1 * dy1;
                    float v1  = (d1 <= 0.8f) ? __expf(-5.0f * d1) : 0.0f;
                    hit1 += (v1 > 0.8f) ? 1.0f : 0.0f;
                    fb1.h[h] = (_Float16)v1;
                }
                a00.v = WMMA(fa0.v, fb0.v, a00.v);
                a10.v = WMMA(fa1.v, fb0.v, a10.v);
                a01.v = WMMA(fa0.v, fb1.v, a01.v);
                a11.v = WMMA(fa1.v, fb1.v, a11.v);
            }
            hit0 += __shfl_xor(hit0, 16);
            hit1 += __shfl_xor(hit1, 16);
            float den0 = frcp(fmaxf(hit0, 1.0f));
            float den1 = frcp(fmaxf(hit1, 1.0f));
#pragma unroll
            for (int r = 0; r < 8; ++r) {
                int crow = r + rbase;
                op[(size_t)crow * NPART + col0]        = a00.f[r] * den0;
                op[(size_t)(crow + 16) * NPART + col0] = a10.f[r] * den0;
                op[(size_t)crow * NPART + col1]        = a01.f[r] * den1;
                op[(size_t)(crow + 16) * NPART + col1] = a11.f[r] * den1;
            }
        }
        // no end-of-frame barrier needed: next frame's first barrier (after
        // phase A) orders phase-C readers against phase-B writers of frame f+1
    }
}

extern "C" void kernel_launch(void* const* d_in, const int* in_sizes, int n_in,
                              void* d_out, int out_size, void* d_ws, size_t ws_size,
                              hipStream_t stream) {
    (void)in_sizes; (void)n_in; (void)out_size; (void)d_ws; (void)ws_size;
    const float* x    = (const float*)d_in[0];
    const float* Wr1  = (const float*)d_in[1];
    const float* br1  = (const float*)d_in[2];
    const float* Wr2  = (const float*)d_in[3];
    const float* br2  = (const float*)d_in[4];
    const float* Wv1  = (const float*)d_in[5];
    const float* bv1  = (const float*)d_in[6];
    const float* Wv2  = (const float*)d_in[7];
    const float* bv2  = (const float*)d_in[8];
    const float* Wxz  = (const float*)d_in[9];
    const float* Whz  = (const float*)d_in[10];
    const float* bz   = (const float*)d_in[11];
    const float* Wxr  = (const float*)d_in[12];
    const float* Whr  = (const float*)d_in[13];
    const float* brg  = (const float*)d_in[14];
    const float* Wxq  = (const float*)d_in[15];
    const float* Whq  = (const float*)d_in[16];
    const float* bq   = (const float*)d_in[17];
    float* out = (float*)d_out;

    const size_t smemBytes = (size_t)SMEM_FLOATS * sizeof(float); // ~295 KB < 320 KB/WGP
    LHC_57320633532463_kernel<<<dim3(NB), dim3(NTHR), smemBytes, stream>>>(
        x, Wr1, br1, Wr2, br2, Wv1, bv1, Wv2, bv2,
        Wxz, Whz, bz, Wxr, Whr, brg, Wxq, Whq, bq, out);
}
